// RPNLayer_77395310673976
// MI455X (gfx1250) — matrix operand order repaired and checked by reading, not compiled, not run on hardware
//
#include <hip/hip_runtime.h>
#include <hip/hip_bf16.h>
#include <math.h>
#include <stdint.h>

// ---------------------------------------------------------------------------
// RPN layer for MI455X (gfx1250). The 3x3 conv (M=10500,N=512,K=9216, ~99
// GFLOP) dominates; implemented as implicit-GEMM with v_wmma_f32_16x16x32_f16
// (f16 in, f32 accumulate). Tile staging uses CDNA5 async global->LDS copies
// (GLOBAL_LOAD_ASYNC_TO_LDS_B128, ASYNCcnt) with double-buffered LDS so each
// K-step needs a single workgroup barrier. Memory traffic ~85 MB => ~3.7us at
// 23.3 TB/s; the conv is matrix-core bound, so f16 WMMA is the right path.
// ---------------------------------------------------------------------------

typedef __attribute__((ext_vector_type(16))) _Float16 v16h;
typedef __attribute__((ext_vector_type(8)))  _Float16 v8h;
typedef __attribute__((ext_vector_type(8)))  float    v8f;

#define FH     84
#define FW     125
#define MPOS   (FH*FW)          // 10500
#define CIN    1024
#define NOUT   512
#define HP     (FH+2)           // 86
#define WP     (FW+2)           // 127
#define KTOT   (CIN*9)          // 9216
#define KSTEPS (KTOT/32)        // 288
#define NANCH  9
#define ROWS   (MPOS*NANCH)     // 94500
#define SORTN  131072           // next pow2 >= ROWS
#define PRE    6000
#define POST   300
#define WORDS  ((PRE+63)/64)    // 94
#define NMSTH  0.7f

// Get wave-relative LDS byte offset from a generic pointer to __shared__.
// (ISA 10.2: flat LDS addresses carry the LDS offset in addr[31:0].)
__device__ __forceinline__ unsigned lds_off(const void* p) {
    return (unsigned)(uintptr_t)p;
}

// -------------------------- pack feature map (pad+HWC, f32->f16) -----------
__global__ void pack_fm_kernel(const float* __restrict__ fm, _Float16* __restrict__ fmh) {
    long long idx = (long long)blockIdx.x * 256 + threadIdx.x;
    const long long total = (long long)HP * WP * CIN;
    if (idx >= total) return;
    int c  = (int)(idx & (CIN - 1));
    int sp = (int)(idx >> 10);
    int x  = sp % WP;
    int y  = sp / WP;
    float v = 0.0f;
    if (y >= 1 && y <= FH && x >= 1 && x <= FW)
        v = fm[((long long)c * FH + (y - 1)) * FW + (x - 1)];
    fmh[idx] = (_Float16)v;
}

// -------------------------- pack weights [N][K], k = r*1024+c --------------
__global__ void pack_w_kernel(const float* __restrict__ w, _Float16* __restrict__ wp) {
    long long idx = (long long)blockIdx.x * 256 + threadIdx.x;
    const long long total = (long long)NOUT * KTOT;
    if (idx >= total) return;
    int n = (int)(idx / KTOT);
    int k = (int)(idx % KTOT);
    int r = k >> 10;            // 3x3 tap index 0..8
    int c = k & (CIN - 1);
    int ky = r / 3, kx = r % 3;
    wp[idx] = (_Float16)w[(((long long)n * CIN + c) * 3 + ky) * 3 + kx];
}

// -------------------------- 3x3 conv as WMMA implicit GEMM -----------------
// Block: 256 threads = 8 waves; tile 64(M) x 128(N); wave -> 32x32 subtile
// (2x2 fragments of 16x16). K loop: 9 taps x 32-channel steps = 288 rounds.
// Double-buffered LDS, async global->LDS staging, 1 barrier per round.
__global__ __launch_bounds__(256) void conv3x3_wmma_kernel(
    const _Float16* __restrict__ fmh,   // [HP][WP][CIN] padded
    const _Float16* __restrict__ wp,    // [NOUT][KTOT]
    const float*    __restrict__ bias,  // [NOUT]
    _Float16*       __restrict__ out)   // [MPOS][NOUT]
{
    __shared__ __align__(32) _Float16 Ash[2][64 * 32];
    __shared__ __align__(32) _Float16 Bsh[2][128 * 32];

    const int tid  = threadIdx.x;
    const int lane = tid & 31;
    const int wave = tid >> 5;
    const int wm   = wave & 1;      // M half (0..1)
    const int wn   = wave >> 1;     // N quarter (0..3)
    const int mBase = blockIdx.x * 64;
    const int nBase = blockIdx.y * 128;

    v8f c00 = {}, c01 = {}, c10 = {}, c11 = {};

    // A-tile loader: 64 rows x 32 halves, 8 halves (16B, one b128)/thread
    const int a_m  = tid >> 2;
    const int a_kc = (tid & 3) * 8;
    const int am   = mBase + a_m;
    const bool amv = am < MPOS;
    const int ay   = amv ? am / FW : 0;
    const int ax   = amv ? am % FW : 0;

    // B-tile loader: 128 cols x 32 halves, 16 halves (32B, two b128)/thread
    const int b_n = tid >> 1;
    const int b_k = (tid & 1) * 16;

    // Out-of-range M rows stay zero forever: fill once in both buffers,
    // async loads for those lanes are EXEC-masked off below.
    if (!amv) {
        v8h z = {};
        *(v8h*)(&Ash[0][a_m * 32 + a_kc]) = z;
        *(v8h*)(&Ash[1][a_m * 32 + a_kc]) = z;
    }

    // Issue async global->LDS staging for K-step `ks` into buffer `buf`.
    auto issue_tile = [&](int ks, int buf) {
        const int r  = ks >> 5;             // 3x3 tap (32 steps per tap)
        const int cb = (ks & 31) << 5;      // channel base within tap
        const int dy = r / 3, dx = r % 3;   // padded spatial offsets
        // A tile
        if (amv) {
            const _Float16* asrc =
                fmh + ((long long)(ay + dy) * WP + (ax + dx)) * CIN + cb + a_kc;
            unsigned adst = lds_off(&Ash[buf][a_m * 32 + a_kc]);
            asm volatile("global_load_async_to_lds_b128 %0, %1, off"
                         :: "v"(adst), "v"(asrc) : "memory");
        }
        // B tile (32 contiguous bytes: two b128; offset applies to both sides)
        {
            const _Float16* bsrc =
                wp + (long long)(nBase + b_n) * KTOT + (r << 10) + cb + b_k;
            unsigned bdst = lds_off(&Bsh[buf][b_n * 32 + b_k]);
            asm volatile("global_load_async_to_lds_b128 %0, %1, off\n\t"
                         "global_load_async_to_lds_b128 %0, %1, off offset:16"
                         :: "v"(bdst), "v"(bsrc) : "memory");
        }
    };

    // Prologue: stage step 0 into buffer 0.
    issue_tile(0, 0);
    asm volatile("s_wait_asynccnt 0x0" ::: "memory");
    __syncthreads();

    const int klo   = (lane >> 4) * 16;     // A: lane<16 -> K 0..15
    const int arow0 = wm * 32 + (lane & 15);
    const int arow1 = arow0 + 16;
    const int bcol0 = wn * 32 + (lane & 15);
    const int bk0   = (lane >> 4) * 16;     // B: lane<16 -> K 0..15 of its col

    for (int s = 0; s < KSTEPS; ++s) {
        const int cur = s & 1;
        if (s + 1 < KSTEPS) issue_tile(s + 1, cur ^ 1);

        const _Float16* Ab = Ash[cur];
        const _Float16* Bb = Bsh[cur];

        v16h a0, a1;
        {
            v8h lo0 = *(const v8h*)(Ab + arow0 * 32 + klo);
            v8h hi0 = *(const v8h*)(Ab + arow0 * 32 + klo + 8);
            v8h lo1 = *(const v8h*)(Ab + arow1 * 32 + klo);
            v8h hi1 = *(const v8h*)(Ab + arow1 * 32 + klo + 8);
#pragma unroll
            for (int q = 0; q < 8; ++q) {
                a0[q] = lo0[q]; a0[q + 8] = hi0[q];
                a1[q] = lo1[q]; a1[q + 8] = hi1[q];
            }
        }
        v16h b0 = *(const v16h*)(Bb + bcol0 * 32 + bk0);
        v16h b1 = *(const v16h*)(Bb + (bcol0 + 16) * 32 + bk0);

        c00 = __builtin_amdgcn_wmma_f32_16x16x32_f16(false, a0, false, b0, (short)0, c00, false, false);
        c01 = __builtin_amdgcn_wmma_f32_16x16x32_f16(false, a0, false, b1, (short)0, c01, false, false);
        c10 = __builtin_amdgcn_wmma_f32_16x16x32_f16(false, a1, false, b0, (short)0, c10, false, false);
        c11 = __builtin_amdgcn_wmma_f32_16x16x32_f16(false, a1, false, b1, (short)0, c11, false, false);

        // Next buffer's async copies must land; then one barrier covers both
        // "next buffer ready" and "current buffer free for step s+2".
        asm volatile("s_wait_asynccnt 0x0" ::: "memory");
        __syncthreads();
    }

    // Epilogue: C/D layout (VGPR i: M=i lanes0-15, M=i+8 lanes16-31; N=lane&15)
    const int mhalf = (lane >> 4) * 8;
    const int ncol  = lane & 15;
    const int ng0 = nBase + wn * 32 + ncol;
    const int ng1 = ng0 + 16;
    const float bs0 = bias[ng0];
    const float bs1 = bias[ng1];
#pragma unroll
    for (int i = 0; i < 8; ++i) {
        const int mg0 = mBase + wm * 32 + i + mhalf;
        const int mg1 = mg0 + 16;
        if (mg0 < MPOS) {
            out[(long long)mg0 * NOUT + ng0] = (_Float16)fmaxf(c00[i] + bs0, 0.0f);
            out[(long long)mg0 * NOUT + ng1] = (_Float16)fmaxf(c01[i] + bs1, 0.0f);
        }
        if (mg1 < MPOS) {
            out[(long long)mg1 * NOUT + ng0] = (_Float16)fmaxf(c10[i] + bs0, 0.0f);
            out[(long long)mg1 * NOUT + ng1] = (_Float16)fmaxf(c11[i] + bs1, 0.0f);
        }
    }
}

// -------------------------- 1x1 heads (cls 18 + reg 36) + softmax ----------
__global__ __launch_bounds__(256) void heads_kernel(
    const _Float16* __restrict__ feat,   // [MPOS][512]
    const float* __restrict__ cls_w, const float* __restrict__ cls_b,
    const float* __restrict__ reg_w, const float* __restrict__ reg_b,
    float* __restrict__ bbox_out,        // [ROWS][4]
    float* __restrict__ cls_out,         // [ROWS][2]
    float* __restrict__ fg)              // [ROWS]
{
    __shared__ _Float16 wsh[54 * 512];
    for (int i = threadIdx.x; i < 54 * 512; i += 256) {
        int j = i >> 9, c = i & 511;
        float v = (j < 18) ? cls_w[j * 512 + c] : reg_w[(j - 18) * 512 + c];
        wsh[i] = (_Float16)v;
    }
    __syncthreads();

    const int m = blockIdx.x * 256 + threadIdx.x;
    if (m >= MPOS) return;

    float acc[54];
#pragma unroll
    for (int j = 0; j < 54; ++j) acc[j] = 0.0f;

    const _Float16* f = feat + (long long)m * 512;
    for (int c = 0; c < 512; ++c) {
        const float fv = (float)f[c];
#pragma unroll
        for (int j = 0; j < 54; ++j) acc[j] += fv * (float)wsh[j * 512 + c];
    }

#pragma unroll
    for (int a = 0; a < 9; ++a) {
        const int row = m * 9 + a;
        const float s0 = acc[2 * a]     + cls_b[2 * a];
        const float s1 = acc[2 * a + 1] + cls_b[2 * a + 1];
        cls_out[row * 2 + 0] = s0;
        cls_out[row * 2 + 1] = s1;
        fg[row] = 1.0f / (1.0f + expf(s0 - s1));
#pragma unroll
        for (int t = 0; t < 4; ++t)
            bbox_out[row * 4 + t] = acc[18 + 4 * a + t] + reg_b[4 * a + t];
    }
}

// -------------------------- anchors ----------------------------------------
__global__ void anchors_kernel(float* __restrict__ anc) {
    const int row = blockIdx.x * 256 + threadIdx.x;
    if (row >= ROWS) return;
    const int m = row / 9, a = row % 9;
    const int y = m / FW, x = m % FW;
    const int ra = a / 3, sa = a % 3;
    const float ratio = (ra == 0) ? 0.5f : ((ra == 1) ? 1.0f : 2.0f);
    const float scale = (sa == 0) ? 8.0f : ((sa == 1) ? 16.0f : 32.0f);
    const float wsr = roundf(sqrtf(256.0f / ratio));
    const float hsr = roundf(wsr * ratio);
    const float ww = wsr * scale, hh = hsr * scale;
    const float ctr = 7.5f;
    const float sx = x * 16.0f, sy = y * 16.0f;
    anc[row * 4 + 0] = sx + ctr - 0.5f * (ww - 1.0f);
    anc[row * 4 + 1] = sy + ctr - 0.5f * (hh - 1.0f);
    anc[row * 4 + 2] = sx + ctr + 0.5f * (ww - 1.0f);
    anc[row * 4 + 3] = sy + ctr + 0.5f * (hh - 1.0f);
}

// -------------------------- decode + clip + filter -> sort keys -------------
__global__ void decode_kernel(
    const float* __restrict__ bbox, const float* __restrict__ anc,
    const float* __restrict__ fg,   const float* __restrict__ im_info,
    float* __restrict__ boxes, float* __restrict__ keys, int* __restrict__ idxs)
{
    const int i = blockIdx.x * 256 + threadIdx.x;
    if (i >= SORTN) return;
    if (i >= ROWS) { keys[i] = -3.0e38f; idxs[i] = 0; return; }
    const float ax1 = anc[i * 4], ay1 = anc[i * 4 + 1];
    const float ax2 = anc[i * 4 + 2], ay2 = anc[i * 4 + 3];
    const float wa = ax2 - ax1 + 1.0f, ha = ay2 - ay1 + 1.0f;
    const float cxa = ax1 + 0.5f * wa, cya = ay1 + 0.5f * ha;
    const float ddx = bbox[i * 4], ddy = bbox[i * 4 + 1];
    const float ddw = bbox[i * 4 + 2], ddh = bbox[i * 4 + 3];
    const float cx = ddx * wa + cxa, cy = ddy * ha + cya;
    const float bw = wa * expf(ddw), bh = ha * expf(ddh);
    const float imH = im_info[0], imW = im_info[1], imS = im_info[2];
    const float x1 = fminf(fmaxf(cx - 0.5f * bw, 0.0f), imW - 1.0f);
    const float y1 = fminf(fmaxf(cy - 0.5f * bh, 0.0f), imH - 1.0f);
    const float x2 = fminf(fmaxf(cx + 0.5f * bw, 0.0f), imW - 1.0f);
    const float y2 = fminf(fmaxf(cy + 0.5f * bh, 0.0f), imH - 1.0f);
    boxes[i * 4 + 0] = x1; boxes[i * 4 + 1] = y1;
    boxes[i * 4 + 2] = x2; boxes[i * 4 + 3] = y2;
    const float ms = 16.0f * imS;
    const bool valid = (x2 - x1 + 1.0f >= ms) && (y2 - y1 + 1.0f >= ms);
    keys[i] = valid ? fg[i] : -1.0f;
    idxs[i] = i;
}

// -------------------------- bitonic sort pass (descending) ------------------
__global__ void bitonic_kernel(float* __restrict__ keys, int* __restrict__ idxs,
                               int k, int j) {
    const int i = blockIdx.x * 256 + threadIdx.x;
    const int p = i ^ j;
    if (p <= i) return;
    const bool up = ((i & k) == 0);          // up => descending block
    const float ki = keys[i], kp = keys[p];
    const bool sw = up ? (ki < kp) : (ki > kp);
    if (sw) {
        keys[i] = kp; keys[p] = ki;
        const int t = idxs[i]; idxs[i] = idxs[p]; idxs[p] = t;
    }
}

// -------------------------- gather top PRE proposals ------------------------
__global__ void gather_kernel(const float* __restrict__ keys, const int* __restrict__ idxs,
                              const float* __restrict__ boxes,
                              float* __restrict__ props, float* __restrict__ topsc) {
    const int i = blockIdx.x * 256 + threadIdx.x;
    if (i >= PRE) return;
    const int id = idxs[i];
    topsc[i] = keys[i];
    props[i * 4 + 0] = boxes[id * 4 + 0];
    props[i * 4 + 1] = boxes[id * 4 + 1];
    props[i * 4 + 2] = boxes[id * 4 + 2];
    props[i * 4 + 3] = boxes[id * 4 + 3];
}

// -------------------------- pairwise IoU bitmask ----------------------------
__global__ __launch_bounds__(64) void iou_kernel(const float* __restrict__ props,
                                                 unsigned long long* __restrict__ mask) {
    __shared__ float cb[64 * 4];
    const int t = threadIdx.x;
    const int col0 = blockIdx.y * 64;
    const int c = col0 + t;
    if (c < PRE) {
        cb[t * 4 + 0] = props[c * 4 + 0];
        cb[t * 4 + 1] = props[c * 4 + 1];
        cb[t * 4 + 2] = props[c * 4 + 2];
        cb[t * 4 + 3] = props[c * 4 + 3];
    } else {
        cb[t * 4 + 0] = 0.0f; cb[t * 4 + 1] = 0.0f;
        cb[t * 4 + 2] = 0.0f; cb[t * 4 + 3] = 0.0f;
    }
    __syncthreads();
    const int row = blockIdx.x * 64 + t;
    if (row >= PRE) return;
    const float x1 = props[row * 4], y1 = props[row * 4 + 1];
    const float x2 = props[row * 4 + 2], y2 = props[row * 4 + 3];
    const float area = (x2 - x1 + 1.0f) * (y2 - y1 + 1.0f);
    unsigned long long bits = 0ull;
    for (int q = 0; q < 64; ++q) {
        const int cc = col0 + q;
        if (cc >= PRE || cc == row) continue;
        const float cx1 = cb[q * 4], cy1 = cb[q * 4 + 1];
        const float cx2 = cb[q * 4 + 2], cy2 = cb[q * 4 + 3];
        const float carea = (cx2 - cx1 + 1.0f) * (cy2 - cy1 + 1.0f);
        float iw = fminf(x2, cx2) - fmaxf(x1, cx1) + 1.0f; if (iw < 0.0f) iw = 0.0f;
        float ih = fminf(y2, cy2) - fmaxf(y1, cy1) + 1.0f; if (ih < 0.0f) ih = 0.0f;
        const float inter = iw * ih;
        const float iou = inter / (area + carea - inter);
        if (iou > NMSTH) bits |= (1ull << q);
    }
    mask[(long long)row * WORDS + blockIdx.y] = bits;
}

// -------------------------- greedy NMS reduce + emit ROIs -------------------
__global__ __launch_bounds__(128) void nms_kernel(
    const unsigned long long* __restrict__ mask,
    const float* __restrict__ topsc, const float* __restrict__ props,
    float* __restrict__ rois) {
    __shared__ unsigned long long remv[WORDS];
    __shared__ int keep[PRE];
    __shared__ int nk;
    const int t = threadIdx.x;
    if (t < WORDS) remv[t] = 0ull;
    if (t == 0) nk = 0;
    __syncthreads();
    for (int i = 0; i < PRE; ++i) {
        const bool alive = ((remv[i >> 6] >> (i & 63)) & 1ull) == 0ull;
        __syncthreads();
        if (alive) {
            if (t == 0 && topsc[i] > 0.0f) keep[nk++] = i;
            if (t < WORDS) remv[t] |= mask[(long long)i * WORDS + t];
        }
        __syncthreads();
    }
    for (int s = t; s < POST; s += 128) {
        const bool ok = s < nk;
        const int id = ok ? keep[s] : 0;
        rois[s * 4 + 0] = ok ? props[id * 4 + 0] : 0.0f;
        rois[s * 4 + 1] = ok ? props[id * 4 + 1] : 0.0f;
        rois[s * 4 + 2] = ok ? props[id * 4 + 2] : 0.0f;
        rois[s * 4 + 3] = ok ? props[id * 4 + 3] : 0.0f;
    }
}

// ---------------------------------------------------------------------------
extern "C" void kernel_launch(void* const* d_in, const int* in_sizes, int n_in,
                              void* d_out, int out_size, void* d_ws, size_t ws_size,
                              hipStream_t stream) {
    const float* fm      = (const float*)d_in[0];
    const float* im_info = (const float*)d_in[1];
    const float* conv_w  = (const float*)d_in[2];
    const float* conv_b  = (const float*)d_in[3];
    const float* cls_w   = (const float*)d_in[4];
    const float* cls_b   = (const float*)d_in[5];
    const float* reg_w   = (const float*)d_in[6];
    const float* reg_b   = (const float*)d_in[7];

    float* out      = (float*)d_out;
    float* bbox_out = out;                        // 378000
    float* cls_out  = out + ROWS * 4;             // 189000 @ 378000
    float* rois_out = out + ROWS * 4 + ROWS * 2;  // 1200   @ 567000
    float* anc_out  = rois_out + POST * 4;        // 378000 @ 568200

    char* wsp = (char*)d_ws;
    auto carve = [&](size_t bytes) {
        char* p = wsp;
        wsp += (bytes + 255) & ~((size_t)255);
        return p;
    };
    _Float16* fmh   = (_Float16*)carve((size_t)HP * WP * CIN * 2);      // 22.4 MB
    _Float16* wpck  = (_Float16*)carve((size_t)NOUT * KTOT * 2);        //  9.4 MB
    _Float16* relu  = (_Float16*)carve((size_t)MPOS * NOUT * 2);        // 10.8 MB
    float*    fg    = (float*)carve((size_t)ROWS * 4);
    float*    boxes = (float*)carve((size_t)ROWS * 4 * 4);
    float*    keys  = (float*)carve((size_t)SORTN * 4);
    int*      idxs  = (int*)carve((size_t)SORTN * 4);
    float*    props = (float*)carve((size_t)PRE * 4 * 4);
    float*    topsc = (float*)carve((size_t)PRE * 4);
    unsigned long long* mask =
        (unsigned long long*)carve((size_t)PRE * WORDS * 8);            //  4.5 MB
    (void)ws_size; (void)n_in; (void)in_sizes; (void)out_size;

    // 1) pack inputs to f16
    {
        const long long tot = (long long)HP * WP * CIN;
        pack_fm_kernel<<<(unsigned)((tot + 255) / 256), 256, 0, stream>>>(fm, fmh);
    }
    {
        const long long tot = (long long)NOUT * KTOT;
        pack_w_kernel<<<(unsigned)((tot + 255) / 256), 256, 0, stream>>>(conv_w, wpck);
    }

    // 2) 3x3 conv + ReLU via WMMA implicit GEMM (async-LDS staged)
    {
        dim3 grid((MPOS + 63) / 64, NOUT / 128);   // 165 x 4
        conv3x3_wmma_kernel<<<grid, 256, 0, stream>>>(fmh, wpck, conv_b, relu);
    }

    // 3) 1x1 heads + softmax fg prob
    heads_kernel<<<(MPOS + 255) / 256, 256, 0, stream>>>(
        relu, cls_w, cls_b, reg_w, reg_b, bbox_out, cls_out, fg);

    // 4) anchors (also an output)
    anchors_kernel<<<(ROWS + 255) / 256, 256, 0, stream>>>(anc_out);

    // 5) decode + clip + min-size filter -> sort keys
    decode_kernel<<<SORTN / 256, 256, 0, stream>>>(
        bbox_out, anc_out, fg, im_info, boxes, keys, idxs);

    // 6) bitonic sort descending by score (17*18/2 = 153 passes)
    for (int k = 2; k <= SORTN; k <<= 1)
        for (int j = k >> 1; j > 0; j >>= 1)
            bitonic_kernel<<<SORTN / 256, 256, 0, stream>>>(keys, idxs, k, j);

    // 7) gather top-6000 proposals
    gather_kernel<<<(PRE + 255) / 256, 256, 0, stream>>>(keys, idxs, boxes, props, topsc);

    // 8) pairwise IoU bitmask (6000 x 6000)
    {
        dim3 grid(WORDS, WORDS);  // 94 x 94 blocks of 64 threads
        iou_kernel<<<grid, 64, 0, stream>>>(props, mask);
    }

    // 9) greedy NMS + emit 300 zero-padded ROIs
    nms_kernel<<<1, 128, 0, stream>>>(mask, topsc, props, rois_out);
}
